// MultiHeadLatentAttention_40089224741085
// MI455X (gfx1250) — compile-verified, matrix operand order checked by reading, not run on hardware
//
#include <hip/hip_runtime.h>

// MLA: B=1, S=2048, HID=4096, H=32, HK=8, D=128, L=32
// All matmuls on v_wmma_f32_16x16x32_f16 (f16 in, f32 accumulate).
// Intermediates (Q, K_up, V_up^T, attn-out) are kept in f16 so attention and
// the output projection load WMMA fragments directly with b128 vector loads.

typedef __attribute__((ext_vector_type(16))) _Float16 v16h;
typedef __attribute__((ext_vector_type(8)))  _Float16 v8h;
typedef __attribute__((ext_vector_type(8)))  float    v8f;
typedef __attribute__((ext_vector_type(4)))  float    v4f;

// A/B fragment layout (16x32, MxK or (KxN)^T): lane (r = lane&15, ph = lane>>4)
// holds K = ph*8 + 0..7 in elements 0..7 and K = 16 + ph*8 + 0..7 in 8..15.
// Caller passes p = &M[row * ld + k0 + ph*8].

__device__ __forceinline__ v16h frag_load(const float* __restrict__ p) {
  v4f x0 = *(const v4f*)(p + 0);
  v4f x1 = *(const v4f*)(p + 4);
  v4f x2 = *(const v4f*)(p + 16);
  v4f x3 = *(const v4f*)(p + 20);
  v16h f;
#pragma unroll
  for (int i = 0; i < 4; ++i) {
    f[i]      = (_Float16)x0[i];
    f[4 + i]  = (_Float16)x1[i];
    f[8 + i]  = (_Float16)x2[i];
    f[12 + i] = (_Float16)x3[i];
  }
  return f;
}

__device__ __forceinline__ v16h frag_load(const _Float16* __restrict__ p) {
  v8h lo = *(const v8h*)(p);
  v8h hi = *(const v8h*)(p + 16);
  return __builtin_shufflevector(lo, hi, 0, 1, 2, 3, 4, 5, 6, 7, 8, 9, 10, 11,
                                 12, 13, 14, 15);
}

// ---------------------------------------------------------------------------
// C = A[M,K] * W[N,K]^T.  A is f32 or f16; W is f32 (read once, cvt in flight).
// OMODE: 0 = f32 row-major, 1 = f16 row-major, 2 = f16 transposed (C^T: N x M).
// Block = 256 threads = 8 waves as 4(m) x 2(n); wave tile = 32x64 (8 WMMAs
// per 32-deep k-step), double-buffered so loads overlap the XDL pipe.
// ---------------------------------------------------------------------------
template <typename TA, int OMODE>
__global__ void __launch_bounds__(256)
wmma_gemm_xwt(const TA* __restrict__ A, const float* __restrict__ W,
              void* __restrict__ Cv, int M, int N, int K, float outScale) {
  const int lane = threadIdx.x & 31;
  const int wave = threadIdx.x >> 5;
  const int r    = lane & 15;
  const int ph   = lane >> 4;
  const int mBase = blockIdx.y * 128 + (wave & 3) * 32;
  const int nBase = blockIdx.x * 128 + (wave >> 2) * 64;

  v8f acc[2][4] = {};

  const TA*    a0 = A + (size_t)(mBase + r) * K + ph * 8;
  const TA*    a1 = a0 + (size_t)16 * K;
  const float* w0 = W + (size_t)(nBase + r) * K + ph * 8;

  v16h ac[2], bc[4];
  ac[0] = frag_load(a0);
  ac[1] = frag_load(a1);
#pragma unroll
  for (int j = 0; j < 4; ++j) bc[j] = frag_load(w0 + (size_t)(j * 16) * K);

  for (int k0 = 32; k0 < K; k0 += 32) {
    // next k-step fragments: independent of the WMMAs below -> stay in flight
    v16h an[2], bn[4];
    an[0] = frag_load(a0 + k0);
    an[1] = frag_load(a1 + k0);
#pragma unroll
    for (int j = 0; j < 4; ++j)
      bn[j] = frag_load(w0 + (size_t)(j * 16) * K + k0);

#pragma unroll
    for (int mi = 0; mi < 2; ++mi)
#pragma unroll
      for (int j = 0; j < 4; ++j)
        acc[mi][j] = __builtin_amdgcn_wmma_f32_16x16x32_f16(
            false, ac[mi], false, bc[j], (short)0, acc[mi][j], false, false);

    ac[0] = an[0];
    ac[1] = an[1];
#pragma unroll
    for (int j = 0; j < 4; ++j) bc[j] = bn[j];
  }
#pragma unroll
  for (int mi = 0; mi < 2; ++mi)
#pragma unroll
    for (int j = 0; j < 4; ++j)
      acc[mi][j] = __builtin_amdgcn_wmma_f32_16x16x32_f16(
          false, ac[mi], false, bc[j], (short)0, acc[mi][j], false, false);

  // C layout: VGPR v, lane (r, ph) -> row = v + 8*ph, col = r (per 16-tile)
  if (OMODE == 0) {
    float* C = (float*)Cv;
#pragma unroll
    for (int mi = 0; mi < 2; ++mi)
#pragma unroll
      for (int j = 0; j < 4; ++j)
#pragma unroll
        for (int v = 0; v < 8; ++v)
          C[(size_t)(mBase + mi * 16 + v + 8 * ph) * N + nBase + j * 16 + r] =
              acc[mi][j][v] * outScale;
  } else if (OMODE == 1) {
    _Float16* C = (_Float16*)Cv;
#pragma unroll
    for (int mi = 0; mi < 2; ++mi)
#pragma unroll
      for (int j = 0; j < 4; ++j)
#pragma unroll
        for (int v = 0; v < 8; ++v)
          C[(size_t)(mBase + mi * 16 + v + 8 * ph) * N + nBase + j * 16 + r] =
              (_Float16)(acc[mi][j][v] * outScale);
  } else {
    _Float16* Ct = (_Float16*)Cv;  // N x M, rows of C^T are contiguous in m
#pragma unroll
    for (int mi = 0; mi < 2; ++mi)
#pragma unroll
      for (int j = 0; j < 4; ++j) {
        v8h hv;
#pragma unroll
        for (int v = 0; v < 8; ++v)
          hv[v] = (_Float16)(acc[mi][j][v] * outScale);
        *(v8h*)(Ct + (size_t)(nBase + j * 16 + r) * M + mBase + mi * 16 +
                8 * ph) = hv;
      }
  }
}

// ---------------------------------------------------------------------------
// Reproduce the reference's scrambled latent reshape:
//   lat = concat([kc_t (HK,S,L), vc_t], -1).reshape(S, HK*2L)
//   k_cmp = lat[:, :256], v_cmp = lat[:, 256:]
// kc/vc are raw GEMM outputs (S x 256); kc_t[h,s,l] = kc[s, h*32+l].
// ---------------------------------------------------------------------------
__global__ void __launch_bounds__(256)
mla_latent_gather(const float* __restrict__ kc, const float* __restrict__ vc,
                  float* __restrict__ k_cmp, float* __restrict__ v_cmp) {
  const int idx   = blockIdx.x * 256 + threadIdx.x;  // < 2048*256
  const int s_new = idx >> 8;
  const int j     = idx & 255;
  {
    int flat = s_new * 512 + j;
    int h = flat >> 17, rem = flat & 131071;
    int s0 = rem >> 6, c = rem & 63;
    k_cmp[idx] = (c < 32) ? kc[s0 * 256 + h * 32 + c]
                          : vc[s0 * 256 + h * 32 + (c - 32)];
  }
  {
    int flat = s_new * 512 + 256 + j;
    int h = flat >> 17, rem = flat & 131071;
    int s0 = rem >> 6, c = rem & 63;
    v_cmp[idx] = (c < 32) ? kc[s0 * 256 + h * 32 + c]
                          : vc[s0 * 256 + h * 32 + (c - 32)];
  }
}

// ---------------------------------------------------------------------------
// Causal flash attention, WMMA for QK^T and PV, all-f16 operands.
// Q: (S, H*D) f16 (pre-scaled by 1/sqrt(D)); Km: (S, HK*D) f16;
// VT: (HK*D, S) f16 (transposed -> PV B-fragments load contiguously);
// O: (S, H*D) f16.  grid = (S/128, H); block = 256 (8 waves, 16 queries each).
// ---------------------------------------------------------------------------
__global__ void __launch_bounds__(256)
mla_flash_attn(const _Float16* __restrict__ Q, const _Float16* __restrict__ Km,
               const _Float16* __restrict__ VT, _Float16* __restrict__ O) {
  constexpr int HD = 4096, HKD = 1024, D = 128, S = 2048;
  __shared__ __align__(16) _Float16 plds[8][512];  // 16x32 P tile per wave

  const int lane = threadIdx.x & 31;
  const int wave = threadIdx.x >> 5;
  const int r    = lane & 15;
  const int ph   = lane >> 4;
  const int h    = blockIdx.y;
  const int hk   = h >> 2;  // n_rep = 4, consecutive repeat
  const int qBase = blockIdx.x * 128 + wave * 16;

  // Q fragments (4 k-chunks of D=128), straight b128 loads
  v16h qf[4];
#pragma unroll
  for (int dk = 0; dk < 4; ++dk)
    qf[dk] = frag_load(Q + (size_t)(qBase + r) * HD + h * D + dk * 32 + ph * 8);

  v8f o[8] = {};
  float mrow[8], lrow[8];
#pragma unroll
  for (int v = 0; v < 8; ++v) { mrow[v] = -3.0e38f; lrow[v] = 0.0f; }

  _Float16* pw = &plds[wave][0];
  const int nkb = (qBase + 16 + 31) >> 5;  // 32-key blocks (causal bound)

  for (int kb = 0; kb < nkb; ++kb) {
    const int k32 = kb * 32;

    // ---- scores: two 16x16 tiles ----
    v8f sc[2];
#pragma unroll
    for (int sub = 0; sub < 2; ++sub) {
      const int keyb = k32 + sub * 16;
      v8f s = {};
#pragma unroll
      for (int dk = 0; dk < 4; ++dk) {
        v16h kf = frag_load(Km + (size_t)(keyb + r) * HKD + hk * D + dk * 32 +
                            ph * 8);
        s = __builtin_amdgcn_wmma_f32_16x16x32_f16(false, qf[dk], false, kf,
                                                   (short)0, s, false, false);
      }
      const int key = keyb + r;
#pragma unroll
      for (int v = 0; v < 8; ++v) {
        int qrow = qBase + v + 8 * ph;
        if (key > qrow) s[v] = -1.0e30f;  // causal mask
      }
      sc[sub] = s;
    }

    // ---- preload all V fragments (contiguous from VT); they stay in flight
    //      across the softmax VALU work and the LDS P relayout ----
    v16h vfr[8];
#pragma unroll
    for (int f = 0; f < 8; ++f)
      vfr[f] = frag_load(VT + (size_t)(hk * 128 + f * 16 + r) * S + k32 +
                         ph * 8);

    // ---- online softmax (rows live per-VGPR per-half) ----
    float p0[8], p1[8], alpha[8];
#pragma unroll
    for (int v = 0; v < 8; ++v) {
      float t = fmaxf(sc[0][v], sc[1][v]);
#pragma unroll
      for (int msk = 1; msk < 16; msk <<= 1)
        t = fmaxf(t, __shfl_xor(t, msk, 32));  // 16-lane half reduction
      float mn = fmaxf(mrow[v], t);
      alpha[v] = __expf(mrow[v] - mn);
      mrow[v]  = mn;
      p0[v] = __expf(sc[0][v] - mn);
      p1[v] = __expf(sc[1][v] - mn);
      float rs = p0[v] + p1[v];
#pragma unroll
      for (int msk = 1; msk < 16; msk <<= 1) rs += __shfl_xor(rs, msk, 32);
      lrow[v] = lrow[v] * alpha[v] + rs;
    }
#pragma unroll
    for (int f = 0; f < 8; ++f)
#pragma unroll
      for (int v = 0; v < 8; ++v) o[f][v] *= alpha[v];

    // ---- P: C-layout -> A-layout via per-wave LDS round trip ----
#pragma unroll
    for (int v = 0; v < 8; ++v) {
      int row = v + 8 * ph;
      pw[row * 32 + r]      = (_Float16)p0[v];
      pw[row * 32 + 16 + r] = (_Float16)p1[v];
    }
    asm volatile("s_wait_dscnt 0" ::: "memory");  // in-wave LDS RAW fence
    v8h plo = *(const v8h*)(pw + r * 32 + ph * 8);
    v8h phi = *(const v8h*)(pw + r * 32 + 16 + ph * 8);
    v16h pf = __builtin_shufflevector(plo, phi, 0, 1, 2, 3, 4, 5, 6, 7, 8, 9,
                                      10, 11, 12, 13, 14, 15);

    // ---- PV ----
#pragma unroll
    for (int f = 0; f < 8; ++f)
      o[f] = __builtin_amdgcn_wmma_f32_16x16x32_f16(false, pf, false, vfr[f],
                                                    (short)0, o[f], false,
                                                    false);
  }

  // ---- normalize and store (f16 row-major) ----
  float invl[8];
#pragma unroll
  for (int v = 0; v < 8; ++v) invl[v] = 1.0f / lrow[v];
#pragma unroll
  for (int f = 0; f < 8; ++f)
#pragma unroll
    for (int v = 0; v < 8; ++v)
      O[(size_t)(qBase + v + 8 * ph) * HD + h * D + f * 16 + r] =
          (_Float16)(o[f][v] * invl[v]);
}

// ---------------------------------------------------------------------------
extern "C" void kernel_launch(void* const* d_in, const int* in_sizes, int n_in,
                              void* d_out, int out_size, void* d_ws,
                              size_t ws_size, hipStream_t stream) {
  (void)in_sizes; (void)n_in; (void)out_size; (void)ws_size;
  const float* hidden  = (const float*)d_in[0];
  const float* Wq      = (const float*)d_in[1];
  const float* Wk_down = (const float*)d_in[2];
  const float* Wv_down = (const float*)d_in[3];
  const float* Wk_up   = (const float*)d_in[4];
  const float* Wv_up   = (const float*)d_in[5];
  const float* Wo      = (const float*)d_in[6];
  float* out = (float*)d_out;

  char* ws = (char*)d_ws;
  _Float16* qh    = (_Float16*)ws; ws += (size_t)2048 * 4096 * 2;  // 16 MB
  float*    kc    = (float*)ws;    ws += (size_t)2048 * 256 * 4;   //  2 MB
  float*    vc    = (float*)ws;    ws += (size_t)2048 * 256 * 4;
  float*    kcmp  = (float*)ws;    ws += (size_t)2048 * 256 * 4;
  float*    vcmp  = (float*)ws;    ws += (size_t)2048 * 256 * 4;
  _Float16* kuph  = (_Float16*)ws; ws += (size_t)2048 * 1024 * 2;  //  4 MB
  _Float16* vupT  = (_Float16*)ws; ws += (size_t)2048 * 1024 * 2;  //  4 MB
  _Float16* attnh = (_Float16*)ws; ws += (size_t)2048 * 4096 * 2;  // 16 MB

  dim3 blk(256);
  const float qscale = 0.08838834764831845f;  // 1/sqrt(128), folded into Q

  // q = (hidden @ Wq.T) * scale                 -> f16 (2048 x 4096)
  wmma_gemm_xwt<float, 1><<<dim3(32, 16), blk, 0, stream>>>(
      hidden, Wq, qh, 2048, 4096, 4096, qscale);
  // kc/vc = hidden @ W*_down.T                  -> f32 (2048 x 256)
  wmma_gemm_xwt<float, 0><<<dim3(2, 16), blk, 0, stream>>>(
      hidden, Wk_down, kc, 2048, 256, 4096, 1.0f);
  wmma_gemm_xwt<float, 0><<<dim3(2, 16), blk, 0, stream>>>(
      hidden, Wv_down, vc, 2048, 256, 4096, 1.0f);
  // scrambled latent reshape/split
  mla_latent_gather<<<dim3(2048), blk, 0, stream>>>(kc, vc, kcmp, vcmp);
  // k_up row-major f16; v_up transposed f16 (d-major, s contiguous)
  wmma_gemm_xwt<float, 1><<<dim3(8, 16), blk, 0, stream>>>(
      kcmp, Wk_up, kuph, 2048, 1024, 256, 1.0f);
  wmma_gemm_xwt<float, 2><<<dim3(8, 16), blk, 0, stream>>>(
      vcmp, Wv_up, vupT, 2048, 1024, 256, 1.0f);
  // causal flash attention                       -> f16 (2048 x 4096)
  mla_flash_attn<<<dim3(16, 32), blk, 0, stream>>>(qh, kuph, vupT, attnh);
  // output projection (f16 A operand)            -> f32 (2048 x 4096)
  wmma_gemm_xwt<_Float16, 0><<<dim3(32, 16), blk, 0, stream>>>(
      attnh, Wo, out, 2048, 4096, 4096, 1.0f);
}
